// MultiheadAttention_67843303408354
// MI455X (gfx1250) — matrix operand order repaired and checked by reading, not compiled, not run on hardware
//
#include <hip/hip_runtime.h>

// ---------------------------------------------------------------------------
// MultiheadAttention for MI455X (gfx1250, wave32, WMMA)
// T=2048, B=4, E=1024 -> H=16 heads, HEAD_DIM=KD=64, BH=64 batch-heads.
// d_out = [out: 2048*4*1024 f32][avg_w: 4*2048*2048 f32]
// ws    = Q bf16 [64][2048][64] | K bf16 [64][2048][64] | V^T bf16 [64][64][2048]
// ---------------------------------------------------------------------------

#define TSEQ 2048
#define BH   64
#define HD   64
#define SROW 2052   // padded LDS row stride (floats) for the 16x2048 score strip

typedef __attribute__((ext_vector_type(16))) __bf16       v16bf;
typedef __attribute__((ext_vector_type(8)))  float        v8f;
typedef __attribute__((ext_vector_type(4)))  unsigned int u32x4;
typedef __attribute__((ext_vector_type(4)))  float        f32x4;

union FragBF {
    v16bf v;
    u32x4 u[2];
};

__device__ __forceinline__ v8f wmma_bf16(v16bf a, v16bf b, v8f c) {
    // (neg_a, A, neg_b, B, c_mod, C, reuse_a, reuse_b)
    return __builtin_amdgcn_wmma_f32_16x16x32_bf16(false, a, false, b,
                                                   (short)0, c, false, false);
}

// ---------------------------------------------------------------------------
// Kernel 0: zero the avg_w region of d_out (harness poisons it with 0xAA).
// Regular-temporal stores on purpose: dirties L2 so the later f32 atomics
// from 16 racing heads stay in-cache.
// ---------------------------------------------------------------------------
__global__ void zero_kernel(f32x4* __restrict__ p, size_t n4) {
    size_t i      = (size_t)blockIdx.x * blockDim.x + threadIdx.x;
    size_t stride = (size_t)gridDim.x * blockDim.x;
    const f32x4 z = {0.f, 0.f, 0.f, 0.f};
    for (; i < n4; i += stride) p[i] = z;
}

// ---------------------------------------------------------------------------
// Kernel 1: fused qkv projection.
// x viewed as [T*BH, 64] rows (contiguous), qkv = x @ W^T + bias; Q scaled.
// Writes Q,K as bf16 [bh][t][d]; V transposed as bf16 [bh][d][t].
// One block = 4 rows; weight (192x64) staged transposed in LDS (48 KB).
// ---------------------------------------------------------------------------
__global__ __launch_bounds__(256) void qkv_proj_kernel(
    const float* __restrict__ x, const float* __restrict__ W,
    const float* __restrict__ bias,
    __bf16* __restrict__ Qb, __bf16* __restrict__ Kb, __bf16* __restrict__ Vt)
{
    __shared__ float wt[64 * 192];   // wt[d][j] = W[j][d]  (conflict-free reads)
    __shared__ float xs[4][64];

    const int tid = threadIdx.x;
    for (int idx = tid; idx < 192 * 64; idx += 256) {
        int j = idx >> 6, d = idx & 63;
        wt[d * 192 + j] = W[idx];
    }
    const int  r = tid >> 6;          // row within block (0..3)
    const int  j = tid & 63;          // output/input column (0..63)
    const long n = (long)blockIdx.x * 4 + r;   // row in [T*BH)
    xs[r][j] = x[n * 64 + j];
    __syncthreads();

    float aq = bias[j], ak = bias[64 + j], av = bias[128 + j];
#pragma unroll 8
    for (int d = 0; d < 64; d++) {
        const float xv = xs[r][d];
        aq = fmaf(xv, wt[d * 192 + j], aq);
        ak = fmaf(xv, wt[d * 192 + 64 + j], ak);
        av = fmaf(xv, wt[d * 192 + 128 + j], av);
    }
    const int t  = (int)(n >> 6);
    const int bh = (int)(n & 63);
    const size_t qk = ((size_t)bh * TSEQ + t) * HD + j;
    Qb[qk] = (__bf16)(aq * 0.125f);                      // HEAD_DIM^-0.5 baked in
    Kb[qk] = (__bf16)ak;
    Vt[((size_t)bh * HD + j) * TSEQ + t] = (__bf16)av;   // transposed for P@V
}

// ---------------------------------------------------------------------------
// Kernel 2: attention. grid = (T/16, BH), block = 128 threads (4 waves).
// Per workgroup: 16 query rows of one head; exact softmax over a 16x2048
// LDS strip; avg_w accumulated with global f32 atomics; normalization (1/l)
// folded into the avg_w scale and the P->bf16 fragment conversion.
// ---------------------------------------------------------------------------
__global__ __launch_bounds__(128) void attn_kernel(
    const __bf16* __restrict__ Qb, const __bf16* __restrict__ Kb,
    const __bf16* __restrict__ Vt,
    const float* __restrict__ out_w, const float* __restrict__ out_b,
    float* __restrict__ out, float* __restrict__ avgw)
{
    extern __shared__ __align__(16) float smem[];
    float* sbuf   = smem;                  // 16 x SROW score/exp strip (128 KB)
    float* redb   = smem + 16 * SROW;      // 16 x 9 reduction scratch
    float* rowmax = redb + 16 * 9;         // 16
    float* rowinv = rowmax + 16;           // 16
    float* obuf   = rowinv + 16;           // 16 x 68 attn-tile staging

    const int tid  = threadIdx.x;
    const int wave = tid >> 5;
    const int lane = tid & 31;
    const int mrow = lane & 15;            // M (or N) index within fragment
    const int hi   = lane >> 4;            // half-wave select
    const int t0   = blockIdx.x * 16;
    const int bh   = blockIdx.y;

    const __bf16* Qh = Qb + (size_t)bh * TSEQ * HD;
    const __bf16* Kh = Kb + (size_t)bh * TSEQ * HD;
    const __bf16* Vh = Vt + (size_t)bh * HD * TSEQ;   // [d][t]

    // ---- Q A-fragments (16x64 split into two 16x32), live for whole block.
    // Lane layout (ISA 7.12.2, 16-bit A 16x32): halves 0..7 -> K = hi*8+0..7,
    // halves 8..15 -> K = 16+hi*8+0..7  => two contiguous 16B loads each.
    FragBF qa0, qa1;
    {
        const __bf16* qr = Qh + (size_t)(t0 + mrow) * HD;
        qa0.u[0] = *(const u32x4*)(qr + hi * 8);
        qa0.u[1] = *(const u32x4*)(qr + 16 + hi * 8);
        qa1.u[0] = *(const u32x4*)(qr + 32 + hi * 8);
        qa1.u[1] = *(const u32x4*)(qr + 48 + hi * 8);
    }

    // ---- Phase 1: S = Q @ K^T  (each wave owns s-tiles st = wave, wave+4, ...)
    for (int st = wave; st < TSEQ / 16; st += 4) {
        const int s0 = st * 16;
        // B-frag (32x16, K=d, N=s): lane n=mrow loads K[s0+n][hi*8 .. +7] and [16+hi*8 .. +7]
        const __bf16* kr = Kh + (size_t)(s0 + mrow) * HD;
        // global_prefetch_b8: tile this wave consumes next iteration (64 rows on)
        __builtin_prefetch((const char*)kr + 8192, 0, 1);
        FragBF kb0, kb1;
        kb0.u[0] = *(const u32x4*)(kr + hi * 8);
        kb0.u[1] = *(const u32x4*)(kr + 16 + hi * 8);
        kb1.u[0] = *(const u32x4*)(kr + 32 + hi * 8);
        kb1.u[1] = *(const u32x4*)(kr + 48 + hi * 8);

        v8f acc = {0.f, 0.f, 0.f, 0.f, 0.f, 0.f, 0.f, 0.f};
        acc = wmma_bf16(qa0.v, kb0.v, acc);
        acc = wmma_bf16(qa1.v, kb1.v, acc);

        // C/D layout: VGPR v = row (v + 8*hi), col = lane&15
#pragma unroll
        for (int v = 0; v < 8; v++)
            sbuf[(v + 8 * hi) * SROW + s0 + mrow] = acc[v];
    }
    __syncthreads();

    // ---- Phase 2: exact softmax stats over 16 x 2048 + avg_w accumulation.
    // 8 threads per row; thread covers columns c8*4 + 32*k (f32x4 chunks), so
    // each step the row's 8 threads touch one contiguous 32-col span.
    {
        const int r  = tid >> 3;        // row 0..15
        const int c8 = tid & 7;
        float* row   = sbuf + r * SROW + c8 * 4;

        f32x4 mx4 = {-3.4e38f, -3.4e38f, -3.4e38f, -3.4e38f};
        for (int k = 0; k < 64; k++) {
            f32x4 p = *(const f32x4*)(row + 32 * k);
#pragma unroll
            for (int i = 0; i < 4; i++) mx4[i] = fmaxf(mx4[i], p[i]);
        }
        redb[r * 9 + c8] = fmaxf(fmaxf(mx4[0], mx4[1]), fmaxf(mx4[2], mx4[3]));
        __syncthreads();
        if (c8 == 0) {
            float m2 = redb[r * 9];
            for (int q = 1; q < 8; q++) m2 = fmaxf(m2, redb[r * 9 + q]);
            rowmax[r] = m2;
        }
        __syncthreads();
        const float rm = rowmax[r];
        f32x4 s4 = {0.f, 0.f, 0.f, 0.f};
        for (int k = 0; k < 64; k++) {
            f32x4 p = *(const f32x4*)(row + 32 * k);
#pragma unroll
            for (int i = 0; i < 4; i++) p[i] = __expf(p[i] - rm);
            *(f32x4*)(row + 32 * k) = p;          // unnormalized exp kept in LDS
#pragma unroll
            for (int i = 0; i < 4; i++) s4[i] += p[i];
        }
        redb[r * 9 + c8] = (s4[0] + s4[1]) + (s4[2] + s4[3]);
        __syncthreads();
        if (c8 == 0) {
            float l = 0.f;
            for (int q = 0; q < 8; q++) l += redb[r * 9 + q];
            rowinv[r] = 1.f / l;
        }
        __syncthreads();
        const float sc = rowinv[r] * 0.0625f;     // 1/l * 1/H
        float* avgrow = avgw + ((size_t)((bh >> 4) * TSEQ + t0 + r)) * TSEQ + c8 * 4;
        for (int k = 0; k < 64; k++) {
            f32x4 p = *(const f32x4*)(row + 32 * k);
#pragma unroll
            for (int i = 0; i < 4; i++)
                atomicAdd(avgrow + 32 * k + i, p[i] * sc);  // raced by 16 heads
        }
    }
    __syncthreads();

    // ---- Phase 3: O = Pnorm @ V  (wave owns d-slice d0..d0+15; K-dim = 2048)
    // Row normalization (1/l) folded into the f32->bf16 conversion.
    const int   d0   = wave * 16;
    const float invm = rowinv[mrow];
    v8f acco = {0.f, 0.f, 0.f, 0.f, 0.f, 0.f, 0.f, 0.f};
    for (int kc = 0; kc < TSEQ / 32; kc++) {
        // A-frag from LDS exps (scale by 1/l, convert to bf16)
        const float* prow = sbuf + mrow * SROW + kc * 32 + hi * 8;
        f32x4 f0 = *(const f32x4*)(prow);
        f32x4 f1 = *(const f32x4*)(prow + 4);
        f32x4 f2 = *(const f32x4*)(prow + 16);
        f32x4 f3 = *(const f32x4*)(prow + 20);
        v16bf pv;
#pragma unroll
        for (int i = 0; i < 4; i++) {
            pv[i]      = (__bf16)(f0[i] * invm);
            pv[4 + i]  = (__bf16)(f1[i] * invm);
            pv[8 + i]  = (__bf16)(f2[i] * invm);
            pv[12 + i] = (__bf16)(f3[i] * invm);
        }
        // B-frag from V^T: lane n=mrow loads Vt[d0+n][kc*32 + hi*8 .. +7] and [kc*32 + 16 + hi*8 .. +7]
        const __bf16* vc = Vh + (size_t)(d0 + mrow) * TSEQ + kc * 32 + hi * 8;
        __builtin_prefetch((const char*)vc + 512, 0, 1);   // 8 kc ahead
        FragBF vb;
        vb.u[0] = *(const u32x4*)(vc);
        vb.u[1] = *(const u32x4*)(vc + 16);
        acco = wmma_bf16(pv, vb.v, acco);
    }
#pragma unroll
    for (int v = 0; v < 8; v++)
        obuf[(v + 8 * hi) * 68 + d0 + mrow] = acco[v];
    __syncthreads();

    // ---- Phase 4: out projection (16x64 tile @ out_w^T), exclusive writes.
    {
        const int rp = tid >> 3;           // attn row 0..15
        const int ib = (tid & 7) * 8;      // output cols i..i+7
        const int b  = bh >> 4, h = bh & 15;
        float* orow = out + ((size_t)(t0 + rp) * 4 + b) * 1024 + h * 64;
        for (int i = ib; i < ib + 8; i++) {
            float acc = out_b[i];
#pragma unroll 8
            for (int d = 0; d < 64; d++)
                acc = fmaf(obuf[rp * 68 + d], out_w[i * 64 + d], acc);
            __builtin_nontemporal_store(acc, &orow[i]);   // write-once stream
        }
    }
}

// ---------------------------------------------------------------------------
extern "C" void kernel_launch(void* const* d_in, const int* in_sizes, int n_in,
                              void* d_out, int out_size, void* d_ws, size_t ws_size,
                              hipStream_t stream) {
    const float* x     = (const float*)d_in[0];   // [2048,4,1024]
    const float* W     = (const float*)d_in[1];   // [192,64]
    const float* bias  = (const float*)d_in[2];   // [192]
    const float* out_w = (const float*)d_in[3];   // [64,64]
    const float* out_b = (const float*)d_in[4];   // [64]

    float* out  = (float*)d_out;                          // 2048*4*1024
    float* avgw = out + (size_t)TSEQ * 4 * 1024;          // 4*2048*2048

    __bf16* Qb = (__bf16*)d_ws;
    __bf16* Kb = Qb + (size_t)BH * TSEQ * HD;
    __bf16* Vt = Kb + (size_t)BH * TSEQ * HD;

    zero_kernel<<<2048, 256, 0, stream>>>((f32x4*)avgw,
                                          ((size_t)4 * TSEQ * TSEQ) / 4);

    qkv_proj_kernel<<<(TSEQ * BH) / 4, 256, 0, stream>>>(x, W, bias, Qb, Kb, Vt);

    const size_t lds_bytes =
        (size_t)(16 * SROW + 16 * 9 + 32 + 16 * 68) * sizeof(float);  // ~136 KB
    attn_kernel<<<dim3(TSEQ / 16, BH), 128, lds_bytes, stream>>>(
        Qb, Kb, Vt, out_w, out_b, out, avgw);
}
